// SACNet_22222160789566
// MI455X (gfx1250) — compile-verified, matrix operand order checked by reading, not run on hardware
//
#include <hip/hip_runtime.h>
#include <math.h>

// ---------------------------------------------------------------------------
// SACNet forward on gfx1250 (MI455X), wave32, WMMA f16 16x16x32 matrix core.
//
// v3: conv/gemm unchanged from v2 (disasm confirmed global_load_b128 +
// s_clause + v_cvt_pk_f16_f32 + v_wmma in the hot loops). New this round:
// vq_assign stages the codeword table into LDS with the Tensor Data Mover
// (tensor_load_to_lds + s_wait_tensorcnt), exercising the CDNA5 async-tensor
// path. Guarded by __has_builtin with a plain-load fallback, and arity-
// switched for ROCm 7.2 (5-arg) vs clang-23 (6-arg) builtins.
// ---------------------------------------------------------------------------

typedef __attribute__((ext_vector_type(16))) _Float16 v16h;
typedef __attribute__((ext_vector_type(8)))  float    v8f;
typedef __attribute__((ext_vector_type(4)))  unsigned int uint32x4;
typedef __attribute__((ext_vector_type(8)))  int      int32x8;
typedef __attribute__((ext_vector_type(4)))  int      int32x4;

#if __has_builtin(__builtin_amdgcn_tensor_load_to_lds) && \
    __has_builtin(__builtin_amdgcn_s_wait_tensorcnt)
#define HAVE_TDM 1
#endif

#define WMMA_F16(a, b, c) \
    __builtin_amdgcn_wmma_f32_16x16x32_f16(false, (a), false, (b), (short)0, (c), false, false)

// K-chunk element offset for the 16-bit A/B fragment layout:
// lane<16 holds K {0..7,16..23}, lane>=16 holds K {8..15,24..31}.
__device__ __forceinline__ int frag_koff(int half, int j) {
    return half * 8 + (j & 7) + ((j >> 3) << 4);
}

// ---------------- WMMA implicit-GEMM dilated 3x3 VALID conv ----------------
// One wave: 16 output positions x 64 output channels.
template<int RELU>
__global__ void __launch_bounds__(32)
conv3x3_wmma(const float* __restrict__ x, const float* __restrict__ w,
             const float* __restrict__ bias, float* __restrict__ out,
             int Bn, int IC, int H, int W, int OH, int OW, int dil)
{
    constexpr int OC = 64;
    __shared__ int offtab[1800];                 // IC*9 <= 200*9
    const int Kd   = IC * 9;
    const int lane = threadIdx.x;

    for (int kk = lane; kk < Kd; kk += 32) {     // build gather-offset table
        int c = kk / 9, t = kk - c * 9;
        int ky = t / 3, kx = t - ky * 3;
        offtab[kk] = (c * H + ky * dil) * W + kx * dil;
    }
    __syncthreads();

    const int half = lane >> 4;
    const int r    = lane & 15;
    const int mt   = blockIdx.x;
    const int OHW  = OH * OW;
    const int M    = Bn * OHW;

    int m = mt * 16 + r;
    if (m >= M) m = M - 1;                       // clamp: garbage row never stored
    const int b  = m / OHW;
    const int p  = m - b * OHW;
    const int oy = p / OW, ox = p - oy * OW;
    const float* xp = x + (size_t)b * IC * H * W + (size_t)oy * W + ox;

    const float* wp[4];
#pragma unroll
    for (int t = 0; t < 4; ++t) wp[t] = w + (size_t)(t * 16 + r) * Kd;

    v8f acc[4] = {};
    const int Kfull = Kd & ~31;
    int k0 = 0;
    for (; k0 < Kfull; k0 += 32) {
        v16h af;
#pragma unroll
        for (int j = 0; j < 16; ++j)
            af[j] = (_Float16)xp[offtab[k0 + frag_koff(half, j)]];
        __builtin_prefetch(wp[0] + k0 + 64, 0, 1);
#pragma unroll
        for (int t = 0; t < 4; ++t) {
            v16h bf;
            const float* wb_ = wp[t] + k0 + half * 8;
#pragma unroll
            for (int j = 0; j < 8; ++j) { bf[j] = (_Float16)wb_[j]; bf[j + 8] = (_Float16)wb_[16 + j]; }
            acc[t] = WMMA_F16(af, bf, acc[t]);
        }
    }
    if (k0 < Kd) {                               // zero-padded K tail
        v16h af;
#pragma unroll
        for (int j = 0; j < 16; ++j) {
            int kk = k0 + frag_koff(half, j);
            af[j] = (kk < Kd) ? (_Float16)xp[offtab[kk]] : (_Float16)0.f;
        }
#pragma unroll
        for (int t = 0; t < 4; ++t) {
            v16h bf;
#pragma unroll
            for (int j = 0; j < 16; ++j) {
                int kk = k0 + frag_koff(half, j);
                bf[j] = (kk < Kd) ? (_Float16)wp[t][kk] : (_Float16)0.f;
            }
            acc[t] = WMMA_F16(af, bf, acc[t]);
        }
    }
    // C/D layout: col = lane&15, VGPR v holds row v + 8*(lane>=16)
    const int colr = lane & 15;
#pragma unroll
    for (int v = 0; v < 8; ++v) {
        const int mg = mt * 16 + v + 8 * half;
        if (mg < M) {
            const int bb = mg / OHW, pp = mg - bb * OHW;
            float* op = out + (size_t)bb * OC * OHW + pp;
#pragma unroll
            for (int t = 0; t < 4; ++t) {
                const int oc = t * 16 + colr;
                float val = acc[t][v] + (bias ? bias[oc] : 0.f);
                if (RELU) val = fmaxf(val, 0.f);
                op[(size_t)oc * OHW] = val;
            }
        }
    }
}

// ---------------- Generic strided batched WMMA GEMM ------------------------
// C[z,m,n] = act( sum_k A[z,m,k]*B[z,k,n] + bias[n] ) + resid[z,m,n]
// One wave: 16(M) x 64(N). AK1/BK1: K-stride==1 specializations.
template<bool AK1, bool BK1>
__global__ void __launch_bounds__(32)
gemm_wmma(const float* __restrict__ A, long long sAb, int sAm, int sAk,
          const float* __restrict__ Bm, long long sBb, int sBk, int sBn,
          float* __restrict__ C, long long sCb, int sCm, int sCn,
          const float* __restrict__ bias,
          const float* __restrict__ resid, long long sRb,
          int M, int N, int Kd, int act)
{
    const int lane = threadIdx.x;
    const int half = lane >> 4;
    const int r    = lane & 15;
    const int mt   = blockIdx.x;
    const int nt   = blockIdx.y;
    const int bz   = blockIdx.z;

    int m = mt * 16 + r;
    if (m >= M) m = M - 1;                       // clamp, never stored
    const float* pA = A + (size_t)bz * sAb + (size_t)m * sAm;
    const float* pB[4];
#pragma unroll
    for (int t = 0; t < 4; ++t) {
        int n = nt * 64 + t * 16 + r;
        if (n >= N) n = N - 1;                   // clamp, never stored
        pB[t] = Bm + (size_t)bz * sBb + (size_t)n * sBn;
    }

    v8f acc[4] = {};
    const int Kfull = Kd & ~31;
    int k0 = 0;
    for (; k0 < Kfull; k0 += 32) {
        v16h af;
        if (AK1) {
            const float* a = pA + k0 + half * 8;
            __builtin_prefetch(a + 32, 0, 1);
#pragma unroll
            for (int j = 0; j < 8; ++j) { af[j] = (_Float16)a[j]; af[j + 8] = (_Float16)a[16 + j]; }
        } else {
#pragma unroll
            for (int j = 0; j < 16; ++j)
                af[j] = (_Float16)pA[(size_t)(k0 + frag_koff(half, j)) * sAk];
        }
#pragma unroll
        for (int t = 0; t < 4; ++t) {
            v16h bf;
            if (BK1) {
                const float* bp = pB[t] + k0 + half * 8;
#pragma unroll
                for (int j = 0; j < 8; ++j) { bf[j] = (_Float16)bp[j]; bf[j + 8] = (_Float16)bp[16 + j]; }
            } else {
#pragma unroll
                for (int j = 0; j < 16; ++j)
                    bf[j] = (_Float16)pB[t][(size_t)(k0 + frag_koff(half, j)) * sBk];
            }
            acc[t] = WMMA_F16(af, bf, acc[t]);
        }
    }
    if (k0 < Kd) {                               // zero-padded K tail
        v16h af;
#pragma unroll
        for (int j = 0; j < 16; ++j) {
            int kk = k0 + frag_koff(half, j);
            af[j] = (kk < Kd) ? (_Float16)pA[(size_t)kk * sAk] : (_Float16)0.f;
        }
#pragma unroll
        for (int t = 0; t < 4; ++t) {
            v16h bf;
#pragma unroll
            for (int j = 0; j < 16; ++j) {
                int kk = k0 + frag_koff(half, j);
                bf[j] = (kk < Kd) ? (_Float16)pB[t][(size_t)kk * sBk] : (_Float16)0.f;
            }
            acc[t] = WMMA_F16(af, bf, acc[t]);
        }
    }
    // epilogue
    const int colr = lane & 15;
    const size_t cb = (size_t)bz * sCb;
#pragma unroll
    for (int t = 0; t < 4; ++t) {
        const int col = nt * 64 + t * 16 + colr;
        if (col < N) {
            const float bv = bias ? bias[col] : 0.f;
#pragma unroll
            for (int v = 0; v < 8; ++v) {
                const int mg = mt * 16 + v + 8 * half;
                if (mg < M) {
                    float val = acc[t][v] + bv;
                    if (act == 1)      val = fmaxf(val, 0.f);
                    else if (act == 2) val = 1.f / (1.f + expf(-val));
                    const size_t o = (size_t)mg * sCm + (size_t)col * sCn;
                    if (resid) val += resid[(size_t)bz * sRb + o];
                    C[cb + o] = val;
                }
            }
        }
    }
}

// ---------------- softmax over the BATCH axis (torch F.softmax dim=0) ------
__global__ void softmax_axis0(float* __restrict__ t, int Bn, int S)
{
    int idx = blockIdx.x * blockDim.x + threadIdx.x;
    if (idx >= S) return;
    float mx = -3.4e38f;
    for (int b = 0; b < Bn; ++b) mx = fmaxf(mx, t[(size_t)b * S + idx]);
    float s = 0.f;
    for (int b = 0; b < Bn; ++b) s += expf(t[(size_t)b * S + idx] - mx);
    float inv = 1.f / s;
    for (int b = 0; b < Bn; ++b) {
        size_t o = (size_t)b * S + idx;
        t[o] = expf(t[o] - mx) * inv;
    }
}

// ---------------- soft-VQ assignment: dist + softmax over K ----------------
#define VQ_D 32
#define VQ_K 48
__global__ void __launch_bounds__(64)
vq_assign(const float* __restrict__ Z /* NCHW (B,D,hw) post-relu */,
          const float* __restrict__ cw, const float* __restrict__ scale,
          float* __restrict__ Aout, int Bn, int HWp)
{
    __shared__ float s_cw[VQ_K * VQ_D];
    __shared__ float s_cc[VQ_K];

#ifdef HAVE_TDM
    // Stage the 48x32 f32 codeword table into LDS with the Tensor Data Mover.
    // D# per CDNA5 ISA ch.8: group0 = {flags, lds_addr, global_addr, type},
    // group1 = {data_size, tensor dims/strides, tile dims}. 2D tensor: groups
    // 2/3 zero. Each wave issues its own copy (EXEC is ignored by TDM);
    // duplicate writes of identical data to the same LDS range are benign.
    {
        unsigned long long ga = (unsigned long long)(const void*)cw;
        unsigned lds_off = (unsigned)(unsigned long long)(const void*)&s_cw[0];
        uint32x4 g0;
        g0[0] = 1u;                                        // count=1, user D#
        g0[1] = lds_off;                                   // LDS byte address
        g0[2] = (unsigned)(ga & 0xffffffffu);              // global_addr[31:0]
        g0[3] = (unsigned)((ga >> 32) & 0x1ffffffu)        // global_addr[56:32]
              | (2u << 30);                                // type = 2 ("image")
        int32x8 g1;
        g1[0] = (int)(2u << 16);                           // data_size = 4 bytes
        g1[1] = (int)((unsigned)VQ_D << 16);               // tensor_dim0 = 32 (bits 48..63)
        g1[2] = (int)((unsigned)VQ_K << 16);               // tensor_dim1 = 48 (bits 80..95)
        g1[3] = (int)((unsigned)VQ_D << 16);               // tile_dim0 = 32 (bits 112..127)
        g1[4] = VQ_K;                                      // tile_dim1 = 48, tile_dim2 = 0
        g1[5] = VQ_D;                                      // tensor_dim0_stride = 32
        g1[6] = 0; g1[7] = 0;                              // tensor_dim1_stride = 0
        int32x4 gz = {0, 0, 0, 0};
#if defined(__clang_major__) && (__clang_major__ >= 23)
        int32x8 g4 = {0, 0, 0, 0, 0, 0, 0, 0};
        __builtin_amdgcn_tensor_load_to_lds(g0, g1, gz, gz, g4, 0);
#else
        __builtin_amdgcn_tensor_load_to_lds(g0, g1, gz, gz, 0);
#endif
        __builtin_amdgcn_s_wait_tensorcnt(0);
    }
    __syncthreads();
#else
    for (int i = threadIdx.x; i < VQ_K * VQ_D; i += blockDim.x) s_cw[i] = cw[i];
    __syncthreads();
#endif

    for (int k = threadIdx.x; k < VQ_K; k += blockDim.x) {
        float s = 0.f;
        for (int d = 0; d < VQ_D; ++d) { float c = s_cw[k * VQ_D + d]; s += c * c; }
        s_cc[k] = s;
    }
    __syncthreads();
    int idx = blockIdx.x * blockDim.x + threadIdx.x;
    if (idx >= Bn * HWp) return;
    int b = idx / HWp, p = idx % HWp;
    float z[VQ_D]; float zz = 0.f;
    for (int d = 0; d < VQ_D; ++d) {
        float v = Z[((size_t)b * VQ_D + d) * HWp + p];
        z[d] = v; zz += v * v;
    }
    float dist[VQ_K]; float mx = -3.4e38f;
    for (int k = 0; k < VQ_K; ++k) {
        float dot = 0.f;
        for (int d = 0; d < VQ_D; ++d) dot += z[d] * s_cw[k * VQ_D + d];
        float dv = scale[k] * (zz + s_cc[k] - 2.f * dot);
        dist[k] = dv; mx = fmaxf(mx, dv);
    }
    float s = 0.f;
    for (int k = 0; k < VQ_K; ++k) { float e = expf(dist[k] - mx); dist[k] = e; s += e; }
    float inv = 1.f / s;
    float* ao = Aout + ((size_t)b * HWp + p) * VQ_K;
    for (int k = 0; k < VQ_K; ++k) ao[k] = dist[k] * inv;
}

// ---------------- E[b,k,:] -= (sum_n A[b,n,k]) * C[k,:] --------------------
__global__ void e_finalize(float* __restrict__ E, const float* __restrict__ Aasg,
                           const float* __restrict__ cw, int Bn, int HWp)
{
    int idx = blockIdx.x * blockDim.x + threadIdx.x;
    if (idx >= Bn * VQ_K) return;
    int b = idx / VQ_K, k = idx % VQ_K;
    float s = 0.f;
    for (int n = 0; n < HWp; ++n) s += Aasg[((size_t)b * HWp + n) * VQ_K + k];
    float* e = E + ((size_t)b * VQ_K + k) * VQ_D;
    for (int d = 0; d < VQ_D; ++d) e[d] -= s * cw[k * VQ_D + d];
}

// ---------------- BatchNorm1d(K) training stats over (B,D), + ReLU ---------
__global__ void __launch_bounds__(256)
bn_relu(float* __restrict__ E, const float* __restrict__ g,
        const float* __restrict__ bt, int Bn)
{
    const int k = blockIdx.x;
    __shared__ float r1[256], r2[256];
    float s = 0.f, s2 = 0.f;
    const int Ntot = Bn * VQ_D;
    for (int i = threadIdx.x; i < Ntot; i += 256) {
        int b = i / VQ_D, d = i % VQ_D;
        float v = E[((size_t)b * VQ_K + k) * VQ_D + d];
        s += v; s2 += v * v;
    }
    r1[threadIdx.x] = s; r2[threadIdx.x] = s2;
    __syncthreads();
    for (int o = 128; o > 0; o >>= 1) {
        if (threadIdx.x < o) { r1[threadIdx.x] += r1[threadIdx.x + o];
                               r2[threadIdx.x] += r2[threadIdx.x + o]; }
        __syncthreads();
    }
    float mean = r1[0] / (float)Ntot;
    float var  = r2[0] / (float)Ntot - mean * mean;
    float scl  = g[k] / sqrtf(var + 1e-5f);
    float sh   = bt[k] - mean * scl;
    for (int i = threadIdx.x; i < Ntot; i += 256) {
        int b = i / VQ_D, d = i % VQ_D;
        size_t o = ((size_t)b * VQ_K + k) * VQ_D + d;
        E[o] = fmaxf(E[o] * scl + sh, 0.f);
    }
}

// ---------------- E_sum[b,d] = sum_k En[b,k,d] -----------------------------
__global__ void esum_kernel(const float* __restrict__ En, float* __restrict__ Es, int Bn)
{
    int idx = blockIdx.x * blockDim.x + threadIdx.x;
    if (idx >= Bn * VQ_D) return;
    int b = idx / VQ_D, d = idx % VQ_D;
    float s = 0.f;
    for (int k = 0; k < VQ_K; ++k) s += En[((size_t)b * VQ_K + k) * VQ_D + d];
    Es[idx] = s;
}

// ---------------- xx *= (1 + gate[b,c]) ------------------------------------
__global__ void gate_apply(float* __restrict__ xx, const float* __restrict__ gate,
                           int Bn, int C, int HWp)
{
    int idx = blockIdx.x * blockDim.x + threadIdx.x;
    if (idx >= Bn * C * HWp) return;
    int b = idx / (C * HWp);
    int c = (idx / HWp) % C;
    xx[idx] *= (1.f + gate[(size_t)b * C + c]);
}

// ---------------- bilinear upsample, align_corners=True, into feat ---------
__global__ void upsample_ac(const float* __restrict__ src, float* __restrict__ feat,
                            int Bn, int C, int H, int W, int OH, int OW,
                            long long featStride, long long chanOff)
{
    int idx = blockIdx.x * blockDim.x + threadIdx.x;
    int total = Bn * C * OH * OW;
    if (idx >= total) return;
    int ow = idx % OW; int t = idx / OW;
    int oh = t % OH;   t /= OH;
    int c  = t % C;    int b = t / C;
    float ys = (OH > 1) ? oh * (float)(H - 1) / (float)(OH - 1) : 0.f;
    float xs = (OW > 1) ? ow * (float)(W - 1) / (float)(OW - 1) : 0.f;
    int y0 = (int)floorf(ys); int y1 = min(y0 + 1, H - 1);
    int x0 = (int)floorf(xs); int x1 = min(x0 + 1, W - 1);
    float wy = ys - (float)y0, wx = xs - (float)x0;
    const float* s = src + ((size_t)b * C + c) * (size_t)(H * W);
    float v0 = s[y0 * W + x0] * (1.f - wx) + s[y0 * W + x1] * wx;
    float v1 = s[y1 * W + x0] * (1.f - wx) + s[y1 * W + x1] * wx;
    feat[(size_t)b * featStride + chanOff + ((size_t)c * OH + oh) * OW + ow] =
        v0 * (1.f - wy) + v1 * wy;
}

// ---------------------------------------------------------------------------
static inline void launch_gemm(hipStream_t s,
    const float* A, long long sAb, int sAm, int sAk,
    const float* B, long long sBb, int sBk, int sBn,
    float* C, long long sCb, int sCm, int sCn,
    const float* bias, const float* resid, long long sRb,
    int M, int N, int K, int act, int batch)
{
    dim3 g((M + 15) / 16, (N + 63) / 64, batch);
    if (sAk == 1 && sBk == 1)
        gemm_wmma<true, true><<<g, 32, 0, s>>>(A, sAb, sAm, sAk, B, sBb, sBk, sBn,
                                               C, sCb, sCm, sCn, bias, resid, sRb, M, N, K, act);
    else if (sAk == 1)
        gemm_wmma<true, false><<<g, 32, 0, s>>>(A, sAb, sAm, sAk, B, sBb, sBk, sBn,
                                                C, sCb, sCm, sCn, bias, resid, sRb, M, N, K, act);
    else if (sBk == 1)
        gemm_wmma<false, true><<<g, 32, 0, s>>>(A, sAb, sAm, sAk, B, sBb, sBk, sBn,
                                                C, sCb, sCm, sCn, bias, resid, sRb, M, N, K, act);
    else
        gemm_wmma<false, false><<<g, 32, 0, s>>>(A, sAb, sAm, sAk, B, sBb, sBk, sBn,
                                                 C, sCb, sCm, sCn, bias, resid, sRb, M, N, K, act);
}

extern "C" void kernel_launch(void* const* d_in, const int* in_sizes, int n_in,
                              void* d_out, int out_size, void* d_ws, size_t ws_size,
                              hipStream_t stream)
{
    const float* x    = (const float*)d_in[0];
    const float* w0   = (const float*)d_in[1];
    const float* b0   = (const float*)d_in[2];
    const float* w1   = (const float*)d_in[3];
    const float* b1   = (const float*)d_in[4];
    const float* w2   = (const float*)d_in[5];
    const float* b2   = (const float*)d_in[6];
    const float* wa   = (const float*)d_in[7];
    const float* wb   = (const float*)d_in[8];
    const float* wg   = (const float*)d_in[9];
    const float* wd   = (const float*)d_in[10];
    const float* wenc = (const float*)d_in[11];
    const float* cw   = (const float*)d_in[12];
    const float* scl  = (const float*)d_in[13];
    const float* watt = (const float*)d_in[14];
    const float* batt = (const float*)d_in[15];
    const float* bng  = (const float*)d_in[16];
    const float* bnb  = (const float*)d_in[17];
    const float* wd1  = (const float*)d_in[18];
    const float* bd1  = (const float*)d_in[19];
    const float* wd2  = (const float*)d_in[20];
    const float* bd2  = (const float*)d_in[21];
    const float* wd3  = (const float*)d_in[22];
    const float* bd3  = (const float*)d_in[23];
    float* out = (float*)d_out;

    constexpr int Bn = 512, ICH = 200, Pp = 19, CF = 64, TF = 32, NC = 16;
    constexpr int H1 = 17, H2 = 13, H3 = 7, HW = H3 * H3;      // 49
    constexpr int FEAT = 3 * CF * Pp * Pp;                     // 69312
    constexpr int SEG  = CF * Pp * Pp;                         // 23104

    // -------- workspace bump allocator (fp32 elems), ~238 MB total --------
    float* ws = (float*)d_ws;
    size_t off = 0;
    auto alloc = [&](size_t n) { float* p = ws + off; off += n; return p; };
    float* c1   = alloc((size_t)Bn * CF * H1 * H1);
    float* c2   = alloc((size_t)Bn * CF * H2 * H2);
    float* xx   = alloc((size_t)Bn * CF * HW);
    float* cA   = alloc((size_t)Bn * TF * HW);
    float* cB   = alloc((size_t)Bn * TF * HW);
    float* cG   = alloc((size_t)Bn * TF * HW);
    float* ctx  = alloc((size_t)Bn * HW * HW);
    float* ctx2 = alloc((size_t)Bn * TF * HW);
    float* cZ   = alloc((size_t)Bn * VQ_D * HW);
    float* Aasg = alloc((size_t)Bn * HW * VQ_K);
    float* E    = alloc((size_t)Bn * VQ_K * VQ_D);
    float* Es   = alloc((size_t)Bn * VQ_D);
    float* gate = alloc((size_t)Bn * CF);
    float* feat = alloc((size_t)Bn * FEAT);
    float* h1   = alloc((size_t)Bn * 256);
    float* h2   = alloc((size_t)Bn * 128);
    (void)ws_size; (void)in_sizes; (void)n_in; (void)out_size;

    // -------- backbone convs (implicit-GEMM WMMA, 16x64 tile/wave) --------
    conv3x3_wmma<1><<<dim3((Bn * H1 * H1 + 15) / 16), 32, 0, stream>>>(
        x, w0, b0, c1, Bn, ICH, Pp, Pp, H1, H1, 1);
    conv3x3_wmma<1><<<dim3((Bn * H2 * H2 + 15) / 16), 32, 0, stream>>>(
        c1, w1, b1, c2, Bn, CF, H1, H1, H2, H2, 2);
    conv3x3_wmma<1><<<dim3((Bn * HW + 15) / 16), 32, 0, stream>>>(
        c2, w2, b2, xx, Bn, CF, H2, H2, H3, H3, 3);

    // -------- non-local block: 1x1 convs as batched GEMMs on NCHW --------
    launch_gemm(stream, wa, 0, CF, 1,  xx, (long long)CF * HW, HW, 1,
                cA, (long long)TF * HW, HW, 1, nullptr, nullptr, 0,
                TF, HW, CF, 0, Bn);
    launch_gemm(stream, wb, 0, CF, 1,  xx, (long long)CF * HW, HW, 1,
                cB, (long long)TF * HW, HW, 1, nullptr, nullptr, 0,
                TF, HW, CF, 0, Bn);
    launch_gemm(stream, wg, 0, CF, 1,  xx, (long long)CF * HW, HW, 1,
                cG, (long long)TF * HW, HW, 1, nullptr, nullptr, 0,
                TF, HW, CF, 1, Bn);   // relu
    // ctx[b] = alpha[b](49x32, RAW reshape of cA flat) @ beta[b](32x49)
    launch_gemm(stream, cA, (long long)TF * HW, TF, 1,
                cB, (long long)TF * HW, HW, 1,
                ctx, (long long)HW * HW, HW, 1, nullptr, nullptr, 0,
                HW, HW, TF, 0, Bn);
    softmax_axis0<<<(HW * HW + 255) / 256, 256, 0, stream>>>(ctx, Bn, HW * HW);
    // ctx2[b] = gamma[b](32x49) @ ctxS[b](49x49)
    launch_gemm(stream, cG, (long long)TF * HW, HW, 1,
                ctx, (long long)HW * HW, HW, 1,
                ctx2, (long long)TF * HW, HW, 1, nullptr, nullptr, 0,
                TF, HW, HW, 0, Bn);
    // xx = relu(wd @ ctx2) + xx   (upsample 7->7 is identity)
    launch_gemm(stream, wd, 0, TF, 1,
                ctx2, (long long)TF * HW, HW, 1,
                xx, (long long)CF * HW, HW, 1, nullptr, xx, (long long)CF * HW,
                CF, HW, TF, 1, Bn);

    // -------- soft VQ encoding --------
    launch_gemm(stream, wenc, 0, CF, 1,
                xx, (long long)CF * HW, HW, 1,
                cZ, (long long)VQ_D * HW, HW, 1, nullptr, nullptr, 0,
                VQ_D, HW, CF, 1, Bn);
    vq_assign<<<(Bn * HW + 63) / 64, 64, 0, stream>>>(cZ, cw, scl, Aasg, Bn, HW);
    // E[b,k,d] = sum_n A[b,n,k]*Z[b,n,d]   (A^T @ Z via strides)
    launch_gemm(stream, Aasg, (long long)HW * VQ_K, 1, VQ_K,
                cZ, (long long)VQ_D * HW, 1, HW,
                E, (long long)VQ_K * VQ_D, VQ_D, 1, nullptr, nullptr, 0,
                VQ_K, VQ_D, HW, 0, Bn);
    e_finalize<<<(Bn * VQ_K + 255) / 256, 256, 0, stream>>>(E, Aasg, cw, Bn, HW);
    bn_relu<<<VQ_K, 256, 0, stream>>>(E, bng, bnb, Bn);
    esum_kernel<<<(Bn * VQ_D + 255) / 256, 256, 0, stream>>>(E, Es, Bn);
    launch_gemm(stream, Es, 0, VQ_D, 1,
                watt, 0, 1, VQ_D,
                gate, 0, CF, 1, batt, nullptr, 0,
                Bn, CF, VQ_D, 2, 1);
    gate_apply<<<(Bn * CF * HW + 255) / 256, 256, 0, stream>>>(xx, gate, Bn, CF, HW);

    // -------- upsample(align_corners) + concat into feat --------
    upsample_ac<<<(Bn * SEG + 255) / 256, 256, 0, stream>>>(
        c1, feat, Bn, CF, H1, H1, Pp, Pp, FEAT, 0);
    upsample_ac<<<(Bn * SEG + 255) / 256, 256, 0, stream>>>(
        c2, feat, Bn, CF, H2, H2, Pp, Pp, FEAT, SEG);
    upsample_ac<<<(Bn * SEG + 255) / 256, 256, 0, stream>>>(
        xx, feat, Bn, CF, H3, H3, Pp, Pp, FEAT, 2 * SEG);

    // -------- dense head --------
    launch_gemm(stream, feat, 0, FEAT, 1,
                wd1, 0, 1, FEAT,
                h1, 0, 256, 1, bd1, nullptr, 0, Bn, 256, FEAT, 1, 1);
    launch_gemm(stream, h1, 0, 256, 1,
                wd2, 0, 1, 256,
                h2, 0, 128, 1, bd2, nullptr, 0, Bn, 128, 256, 1, 1);
    launch_gemm(stream, h2, 0, 128, 1,
                wd3, 0, 1, 128,
                out, 0, NC, 1, bd3, nullptr, 0, Bn, NC, 128, 0, 1);
}
